// MoELayer_28527172780239
// MI455X (gfx1250) — matrix operand order repaired and checked by reading, not compiled, not run on hardware
//
#include <hip/hip_runtime.h>

#define T_TOK 8192
#define K_IN  1024
#define O_OUT 1024
#define N_EXP 8

#define TM 128
#define TN 128
#define KC 32
#define NKB (K_IN / KC)   // 32 K-chunks
#define LDSS 40           // 40 bf16 = 80 B row pitch -> conflict-free b128 frag reads

typedef __attribute__((ext_vector_type(16))) __bf16 v16bf;
typedef __attribute__((ext_vector_type(8)))  float  v8f;
typedef __attribute__((ext_vector_type(4)))  int    v4i;

#if __has_builtin(__builtin_amdgcn_global_load_async_to_lds_b128)
#define USE_ASYNC_LDS 1
typedef __attribute__((address_space(1))) v4i gv4i;   // global 16B vector
typedef __attribute__((address_space(3))) v4i lv4i;   // LDS 16B vector
#endif

// 16-byte global -> LDS copy, async (ASYNCcnt-tracked) when available.
__device__ __forceinline__ void g2l_16B(void* lds, const void* g) {
#ifdef USE_ASYNC_LDS
    __builtin_amdgcn_global_load_async_to_lds_b128((gv4i*)g, (lv4i*)lds, 0, 0);
#else
    *(uint4*)lds = *(const uint4*)g;
#endif
}
__device__ __forceinline__ void async_fence() {
#ifdef USE_ASYNC_LDS
    __builtin_amdgcn_s_wait_asynccnt(0);
#endif
}

// ---------------------------------------------------------------------------
// Elementwise f32 -> bf16 (RNE) conversion (used for W_experts and x).
// ---------------------------------------------------------------------------
__global__ __launch_bounds__(256) void convert_bf16_kernel(
    const float* __restrict__ S, __bf16* __restrict__ D) {
    const size_t i = (size_t)(blockIdx.x * 256 + threadIdx.x) * 8;
    float4 a = ((const float4*)(S + i))[0];
    float4 b = ((const float4*)(S + i))[1];
    union { __bf16 h[8]; uint4 q; } o;
    o.h[0] = (__bf16)a.x; o.h[1] = (__bf16)a.y;
    o.h[2] = (__bf16)a.z; o.h[3] = (__bf16)a.w;
    o.h[4] = (__bf16)b.x; o.h[5] = (__bf16)b.y;
    o.h[6] = (__bf16)b.z; o.h[7] = (__bf16)b.w;
    *(uint4*)(D + i) = o.q;
}

// ---------------------------------------------------------------------------
// Router: logits = x . Wr^T + br, softmax, top-2 scatter (not renormalized).
// ---------------------------------------------------------------------------
__global__ __launch_bounds__(256) void moe_router_kernel(
    const float* __restrict__ x, const float* __restrict__ Wr,
    const float* __restrict__ br, float* __restrict__ rw) {
    __shared__ float wr_lds[N_EXP * K_IN];
    const float4* Wr4 = (const float4*)Wr;
    float4* wl4 = (float4*)wr_lds;
    for (int i = threadIdx.x; i < N_EXP * K_IN / 4; i += 256) wl4[i] = Wr4[i];
    __syncthreads();

    const int t = blockIdx.x * 256 + threadIdx.x;
    const float4* xr = (const float4*)(x + (size_t)t * K_IN);
    float acc[N_EXP];
#pragma unroll
    for (int e = 0; e < N_EXP; ++e) acc[e] = 0.f;
    for (int k = 0; k < K_IN / 4; ++k) {
        float4 xv = xr[k];
#pragma unroll
        for (int e = 0; e < N_EXP; ++e) {
            float4 wv = wl4[e * (K_IN / 4) + k];
            acc[e] += xv.x * wv.x + xv.y * wv.y + xv.z * wv.z + xv.w * wv.w;
        }
    }
    float mx = -1e30f;
#pragma unroll
    for (int e = 0; e < N_EXP; ++e) { acc[e] += br[e]; mx = fmaxf(mx, acc[e]); }
    float den = 0.f;
#pragma unroll
    for (int e = 0; e < N_EXP; ++e) { acc[e] = __expf(acc[e] - mx); den += acc[e]; }
    const float inv = 1.f / den;
    int i1 = 0;
#pragma unroll
    for (int e = 1; e < N_EXP; ++e) if (acc[e] > acc[i1]) i1 = e;
    int i2 = (i1 == 0) ? 1 : 0;
#pragma unroll
    for (int e = 0; e < N_EXP; ++e) if (e != i1 && acc[e] > acc[i2]) i2 = e;
#pragma unroll
    for (int e = 0; e < N_EXP; ++e)
        rw[t * N_EXP + e] = (e == i1 || e == i2) ? acc[e] * inv : 0.f;
}

// ---------------------------------------------------------------------------
// Expert GEMM: out[t,o] = sum_e rw[t,e] * (x[t,:].W_e[o,:]) + sum_e rw[t,e]*b_e[o]
// A (x, bf16) staged once per K-chunk via async global->LDS; per-expert router
// weight applied as a per-lane scalar on the A fragment (v_pk_mul_bf16).
// Bias folded in as one extra rank-8 WMMA chunk. Double-buffered LDS.
// __launch_bounds__(256, 2): cap VGPRs < 256 (no s_set_vgpr_msb, 2 waves/SIMD).
// ---------------------------------------------------------------------------
__global__ __launch_bounds__(256, 2) void moe_expert_gemm_kernel(
    const __bf16* __restrict__ xb, const __bf16* __restrict__ Wb,
    const float* __restrict__ be, const float* __restrict__ rw,
    float* __restrict__ out) {
    __shared__ __align__(16) unsigned short As[2][TM * LDSS];
    __shared__ __align__(16) unsigned short Bs[2][TN * LDSS];
    __shared__ float rws[TM * N_EXP];

    const int t0 = blockIdx.x * TM;
    const int n0 = blockIdx.y * TN;
    const int tid  = threadIdx.x;
    const int lane = tid & 31;
    const int wave = tid >> 5;
    const int wm = wave >> 1;      // 0..3 : 32-row strip
    const int wn = wave & 1;       // 0..1 : 64-col strip
    const int srow  = tid >> 1;    // staging row 0..127
    const int shalf = tid & 1;     // staging k-half (0 / 16 elements)
    const int lane_m  = lane & 15;
    const int lane_kh = lane >> 4;

    for (int i = tid; i < TM * N_EXP; i += 256) rws[i] = rw[t0 * N_EXP + i];
    __syncthreads();

    // per-lane router-weight scale factors (a fragment lane = one token row)
    __bf16 swb[2][N_EXP];
#pragma unroll
    for (int mt = 0; mt < 2; ++mt)
#pragma unroll
        for (int e = 0; e < N_EXP; ++e)
            swb[mt][e] = (__bf16)rws[(wm * 32 + mt * 16 + lane_m) * N_EXP + e];

    v8f acc[2][4];
#pragma unroll
    for (int mt = 0; mt < 2; ++mt)
#pragma unroll
        for (int nt = 0; nt < 4; ++nt) { v8f z = {}; acc[mt][nt] = z; }

    auto stageA = [&](int kb, int b) {
        const __bf16* src = xb + (size_t)(t0 + srow) * K_IN + kb * KC + shalf * 16;
        unsigned short* dst = &As[b][srow * LDSS + shalf * 16];
        g2l_16B(dst, src);
        g2l_16B(dst + 8, src + 8);
    };
    auto stageB = [&](int kb, int e, int b) {
        const __bf16* src = Wb + (size_t)e * O_OUT * K_IN
                          + (size_t)(n0 + srow) * K_IN + kb * KC + shalf * 16;
        unsigned short* dst = &Bs[b][srow * LDSS + shalf * 16];
        g2l_16B(dst, src);
        g2l_16B(dst + 8, src + 8);
    };
    // rank-8 bias chunk: A = rw[t,e] (k=e), B = b_e[n] (k=e); zeros elsewhere
    auto stage_bias = [&]() {
        union { __bf16 h[16]; uint4 q[2]; } pa, pb;
#pragma unroll
        for (int i = 0; i < 16; ++i) { pa.h[i] = (__bf16)0.f; pb.h[i] = (__bf16)0.f; }
        if (shalf == 0) {
#pragma unroll
            for (int e = 0; e < N_EXP; ++e) {
                pa.h[e] = (__bf16)rws[srow * N_EXP + e];
                pb.h[e] = (__bf16)be[e * O_OUT + n0 + srow];
            }
        }
        uint4* ad = (uint4*)&As[0][srow * LDSS + shalf * 16];
        ad[0] = pa.q[0]; ad[1] = pa.q[1];
        uint4* bd = (uint4*)&Bs[0][srow * LDSS + shalf * 16];
        bd[0] = pb.q[0]; bd[1] = pb.q[1];
    };

    union Frag { uint4 q[2]; v16bf v; };
    auto loadA = [&](Frag (&a)[2], int b) {
#pragma unroll
        for (int mt = 0; mt < 2; ++mt) {
            const unsigned short* p = &As[b][(wm * 32 + mt * 16 + lane_m) * LDSS];
            a[mt].q[0] = *(const uint4*)(p + lane_kh * 8);       // K 0..7 / 8..15
            a[mt].q[1] = *(const uint4*)(p + 16 + lane_kh * 8);  // K 16..23 / 24..31
        }
    };
    auto compute = [&](const Frag (&a)[2], int b, __bf16 s0, __bf16 s1) {
        Frag as[2];
        as[0].v = a[0].v * s0;       // v_pk_mul_bf16, per-lane scalar scale
        as[1].v = a[1].v * s1;
        Frag bf[4];
#pragma unroll
        for (int nt = 0; nt < 4; ++nt) {
            const unsigned short* p =
                &Bs[b][(wn * 64 + nt * 16 + lane_m) * LDSS];
            bf[nt].q[0] = *(const uint4*)(p + lane_kh * 8);
            bf[nt].q[1] = *(const uint4*)(p + 16 + lane_kh * 8);
        }
#pragma unroll
        for (int mt = 0; mt < 2; ++mt)
#pragma unroll
            for (int nt = 0; nt < 4; ++nt)
                acc[mt][nt] = __builtin_amdgcn_wmma_f32_16x16x32_bf16(
                    false, as[mt].v, false, bf[nt].v,
                    (short)0, acc[mt][nt], false, false);
    };

    // prologue: first A chunk + first B chunk
    stageA(0, 0);
    stageB(0, 0, 0);
    async_fence();
    __syncthreads();

    for (int kb = 0; kb < NKB; ++kb) {
        Frag a[2];
        loadA(a, kb & 1);            // A fragments hoisted: loaded once per kb
#pragma unroll
        for (int e = 0; e < N_EXP; ++e) {
            if (e < N_EXP - 1) {
                stageB(kb, e + 1, (e + 1) & 1);
            } else if (kb < NKB - 1) {
                stageB(kb + 1, 0, 0);
                stageA(kb + 1, (kb + 1) & 1);
            } else {
                stage_bias();        // final chunk: bias rank-8 GEMM
            }
            compute(a, e & 1, swb[0][e], swb[1][e]);
            async_fence();
            __syncthreads();
        }
    }
    {   // bias chunk, unscaled
        Frag a[2];
        loadA(a, 0);
        compute(a, 0, (__bf16)1.0f, (__bf16)1.0f);
    }

    // ---- epilogue: pure stores ----
#pragma unroll
    for (int mt = 0; mt < 2; ++mt) {
#pragma unroll
        for (int nt = 0; nt < 4; ++nt) {
            const int col  = n0 + wn * 64 + nt * 16 + lane_m;
            const int lrow = wm * 32 + mt * 16 + lane_kh * 8;
#pragma unroll
            for (int r = 0; r < 8; ++r)
                out[(size_t)(t0 + lrow + r) * O_OUT + col] = acc[mt][nt][r];
        }
    }
}

extern "C" void kernel_launch(void* const* d_in, const int* in_sizes, int n_in,
                              void* d_out, int out_size, void* d_ws, size_t ws_size,
                              hipStream_t stream) {
    const float* x  = (const float*)d_in[0];
    const float* We = (const float*)d_in[1];
    const float* be = (const float*)d_in[2];
    const float* Wr = (const float*)d_in[3];
    const float* br = (const float*)d_in[4];
    float* out = (float*)d_out;

    const size_t NW = (size_t)N_EXP * O_OUT * K_IN;   // 8M elements
    const size_t NX = (size_t)T_TOK * K_IN;           // 8M elements
    __bf16* Wb = (__bf16*)d_ws;                                  // 16 MB
    __bf16* xb = (__bf16*)((char*)d_ws + NW * sizeof(__bf16));   // 16 MB
    float*  rw = (float*)((char*)d_ws + (NW + NX) * sizeof(__bf16)); // 256 KB

    convert_bf16_kernel<<<(int)(NW / 8 / 256), 256, 0, stream>>>(We, Wb);
    convert_bf16_kernel<<<(int)(NX / 8 / 256), 256, 0, stream>>>(x, xb);
    moe_router_kernel<<<T_TOK / 256, 256, 0, stream>>>(x, Wr, br, rw);
    moe_expert_gemm_kernel<<<dim3(T_TOK / TM, O_OUT / TN), 256, 0, stream>>>(
        xb, Wb, be, rw, out);
}
